// GATModel_39230231281895
// MI455X (gfx1250) — compile-verified, matrix operand order checked by reading, not run on hardware
//
#include <hip/hip_runtime.h>
#include <hip/hip_bf16.h>

// ---------------------------------------------------------------------------
// CDNA5 (gfx1250, wave32) implementation of a 3-layer graph-attention network.
// All 256-K GEMMs run on v_wmma_f32_16x16x32_f16 (f16 in, f32 accumulate).
// The dominant edge-projection GEMM (320000x256 @ 256x256, x3 layers) is fused
// with the per-edge attention: msg tiles are produced in LDS and consumed by
// WMMA directly, then scattered with global f32 atomics (segment_sum).
// ---------------------------------------------------------------------------

typedef _Float16 half_t;
typedef __attribute__((ext_vector_type(16))) _Float16 v16h;
typedef __attribute__((ext_vector_type(8)))  _Float16 v8h;
typedef __attribute__((ext_vector_type(2)))  _Float16 v2h;
typedef __attribute__((ext_vector_type(8)))  float    v8f;

#define HIDN  256
#define NHEAD 4
#define DHEAD 64
#define NNODE 20000
#define NEDGE 320000

// Load one 16x32 f16 A/B WMMA fragment row/col for this lane.
// ISA layout (16-bit A 16x32): lane<16 -> K {0..7, 16..23}; lane>=16 -> K {8..15, 24..31}.
// B mirrors this with N in place of M, so the same loader works with rowbase =
// &W[n*256] (proj = A @ W^T means B[k][n] = W[n][k], contiguous in k).
__device__ __forceinline__ v16h load_frag16(const half_t* __restrict__ rowbase,
                                            int kb, int lane) {
  const int koff = (lane & 16) ? 8 : 0;
  v8h lo = *(const v8h*)(rowbase + kb + koff);
  v8h hi = *(const v8h*)(rowbase + kb + koff + 16);
  v16h r;
#pragma unroll
  for (int i = 0; i < 8; ++i) { r[i] = lo[i]; r[i + 8] = hi[i]; }
  return r;
}

__global__ void k_f32_to_f16(const float* __restrict__ in,
                             half_t* __restrict__ out, int n) {
  int i = blockIdx.x * blockDim.x + threadIdx.x;
  if (i < n) out[i] = (half_t)in[i];
}

__global__ void k_zero4(float* __restrict__ p, int n) {
  int i = (blockIdx.x * blockDim.x + threadIdx.x) * 4;
  if (i < n) { float4 z{0.f, 0.f, 0.f, 0.f}; *(float4*)(p + i) = z; }
}

// ---------------------------------------------------------------------------
// Dense GEMM:  out[M x 256] = A16[M x 256] @ W16^T[256 x 256] + bias
// wave tile 16M x 64N (4 wmma accumulators), block = 8 waves = 32M x 256N.
// MODE 0: f32 out + f16 mirror    (hidden = x @ Win^T + b_in)
// MODE 1: f16 out                 (Q/K/V)
// MODE 2: relu -> f16 out         (t1 = relu(h @ W1^T + b1))
// MODE 3: f32 out = v + resid     (final: x + t1 @ W2^T + b2)
// ---------------------------------------------------------------------------
template <int MODE>
__global__ void __launch_bounds__(256) k_gemm(
    const half_t* __restrict__ A, const half_t* __restrict__ W,
    const float* __restrict__ bias, const float* __restrict__ resid,
    float* __restrict__ outF, half_t* __restrict__ outH) {
  const int lane = threadIdx.x & 31;
  const int wid  = threadIdx.x >> 5;
  const int wm   = wid >> 2;            // 0..1  (M direction)
  const int wn   = wid & 3;             // 0..3  (N direction)
  const int mbase = blockIdx.x * 32 + wm * 16;
  const int nbase = wn * 64;
  const half_t* arow = A + (size_t)(mbase + (lane & 15)) * HIDN;

  v8f acc[4] = {};
#pragma unroll
  for (int kb = 0; kb < HIDN; kb += 32) {
    v16h a = load_frag16(arow, kb, lane);
#pragma unroll
    for (int nt = 0; nt < 4; ++nt) {
      v16h b = load_frag16(W + (size_t)(nbase + nt * 16 + (lane & 15)) * HIDN,
                           kb, lane);
      acc[nt] = __builtin_amdgcn_wmma_f32_16x16x32_f16(
          false, a, false, b, (short)0, acc[nt], false, false);
    }
  }

  // C/D layout: lane holds column (lane&15); rows r(+8 for lanes 16..31).
  const int colrel = lane & 15;
  const int rofs   = (lane & 16) ? 8 : 0;
#pragma unroll
  for (int nt = 0; nt < 4; ++nt) {
    const int col = nbase + nt * 16 + colrel;
    const float bvl = bias[col];
#pragma unroll
    for (int r = 0; r < 8; ++r) {
      const size_t idx = (size_t)(mbase + rofs + r) * HIDN + col;
      float v = acc[nt][r] + bvl;
      if (MODE == 0)      { outF[idx] = v; outH[idx] = (half_t)v; }
      else if (MODE == 1) { outH[idx] = (half_t)v; }
      else if (MODE == 2) { v = v > 0.f ? v : 0.f; outH[idx] = (half_t)v; }
      else                { outF[idx] = v + resid[idx]; }
    }
  }
}

// ---------------------------------------------------------------------------
// Fused per-edge kernel: 128 threads (4 waves), 16 edges per block.
// Phase 1 (wave-per-edge): gather Q[src],K[tgt],V[tgt] (f16), 4x4 attention
//   scores via 32-lane butterfly reduction, softmax over target heads,
//   msg = attn @ V + Eemb[et]  -> LDS tile [16 x 256] f16.
// Phase 2: each wave: 16x64 WMMA tile of proj = msg @ Wo^T + bo, then
//   atomicAdd scatter into agg[src] (segment_sum).
// ---------------------------------------------------------------------------
__global__ void __launch_bounds__(128) k_edge(
    const half_t* __restrict__ Q, const half_t* __restrict__ K,
    const half_t* __restrict__ V, const int* __restrict__ edges,
    const int* __restrict__ etypes, const float* __restrict__ Ee,
    const half_t* __restrict__ Wo, const float* __restrict__ bo,
    float* __restrict__ agg) {
  __shared__ __align__(16) half_t smsg[16 * HIDN];
  __shared__ int ssrc[16];

  const int lane = threadIdx.x & 31;
  const int wid  = threadIdx.x >> 5;
  const int d    = lane * 2;  // this lane covers dims {2*lane, 2*lane+1} of every head

#pragma unroll
  for (int i = 0; i < 4; ++i) {
    const int le = wid * 4 + i;
    const int e  = blockIdx.x * 16 + le;
    const int s  = edges[2 * e];
    const int t  = edges[2 * e + 1];
    const int et = etypes[e];
    if (lane == 0) ssrc[le] = s;
    const half_t* qr = Q + (size_t)s * HIDN;
    const half_t* kr = K + (size_t)t * HIDN;
    const half_t* vr = V + (size_t)t * HIDN;

    float qx[4], qy[4], kx[4], ky[4], vx[4], vy[4];
#pragma unroll
    for (int h = 0; h < 4; ++h) {
      v2h q2 = *(const v2h*)(qr + h * DHEAD + d);
      v2h k2 = *(const v2h*)(kr + h * DHEAD + d);
      v2h w2 = *(const v2h*)(vr + h * DHEAD + d);
      qx[h] = (float)q2[0]; qy[h] = (float)q2[1];
      kx[h] = (float)k2[0]; ky[h] = (float)k2[1];
      vx[h] = (float)w2[0]; vy[h] = (float)w2[1];
    }

    // scores[h][g] = sum_d Q[h,d]*K[g,d] ; each lane holds 2 dims -> reduce.
    float sc[16];
#pragma unroll
    for (int h = 0; h < 4; ++h)
#pragma unroll
      for (int g = 0; g < 4; ++g)
        sc[h * 4 + g] = qx[h] * kx[g] + qy[h] * ky[g];
#pragma unroll
    for (int off = 16; off > 0; off >>= 1)
#pragma unroll
      for (int j = 0; j < 16; ++j)
        sc[j] += __shfl_xor(sc[j], off, 32);

#pragma unroll
    for (int h = 0; h < 4; ++h) {
      const float t0 = sc[h * 4 + 0] * 0.125f;   // 1/sqrt(64)
      const float t1 = sc[h * 4 + 1] * 0.125f;
      const float t2 = sc[h * 4 + 2] * 0.125f;
      const float t3 = sc[h * 4 + 3] * 0.125f;
      const float mx = fmaxf(fmaxf(t0, t1), fmaxf(t2, t3));
      const float e0 = __expf(t0 - mx), e1 = __expf(t1 - mx);
      const float e2 = __expf(t2 - mx), e3 = __expf(t3 - mx);
      const float inv = 1.f / (e0 + e1 + e2 + e3);
      const float p0 = e0 * inv, p1 = e1 * inv, p2 = e2 * inv, p3 = e3 * inv;
      float mx_ = p0 * vx[0] + p1 * vx[1] + p2 * vx[2] + p3 * vx[3];
      float my_ = p0 * vy[0] + p1 * vy[1] + p2 * vy[2] + p3 * vy[3];
      const float* eb = Ee + (size_t)et * HIDN + h * DHEAD + d;
      mx_ += eb[0];
      my_ += eb[1];
      v2h st; st[0] = (half_t)mx_; st[1] = (half_t)my_;
      *(v2h*)(smsg + le * HIDN + h * DHEAD + d) = st;
    }
  }
  __syncthreads();

  // Phase 2: WMMA projection of the 16-edge msg tile, scatter to agg[src].
  const int nbase = wid * 64;
  const half_t* arow = smsg + (lane & 15) * HIDN;
  v8f acc[4] = {};
#pragma unroll
  for (int kb = 0; kb < HIDN; kb += 32) {
    v16h a = load_frag16(arow, kb, lane);
#pragma unroll
    for (int nt = 0; nt < 4; ++nt) {
      v16h b = load_frag16(Wo + (size_t)(nbase + nt * 16 + (lane & 15)) * HIDN,
                           kb, lane);
      acc[nt] = __builtin_amdgcn_wmma_f32_16x16x32_f16(
          false, a, false, b, (short)0, acc[nt], false, false);
    }
  }
  const int colrel = lane & 15;
  const int rofs   = (lane & 16) ? 8 : 0;
#pragma unroll
  for (int nt = 0; nt < 4; ++nt) {
    const int col = nbase + nt * 16 + colrel;
    const float bvl = bo[col];
#pragma unroll
    for (int r = 0; r < 8; ++r) {
      const int node = ssrc[rofs + r];
      atomicAdd(agg + (size_t)node * HIDN + col, acc[nt][r] + bvl);
    }
  }
}

// ---------------------------------------------------------------------------
// hidden = relu(LayerNorm(hidden + agg)) ; also writes f16 mirror.
// One wave per 256-float row; 8 rows per block.
// ---------------------------------------------------------------------------
__global__ void __launch_bounds__(256) k_ln_relu(
    const float* __restrict__ agg, const float* __restrict__ g,
    const float* __restrict__ bta, float* __restrict__ hidden,
    half_t* __restrict__ h16) {
  const int lane = threadIdx.x & 31;
  const int wid  = threadIdx.x >> 5;
  const size_t base = (size_t)(blockIdx.x * 8 + wid) * HIDN;

  float v[8];
  float s = 0.f;
#pragma unroll
  for (int j = 0; j < 8; ++j) {
    const int c = lane + j * 32;
    v[j] = hidden[base + c] + agg[base + c];
    s += v[j];
  }
#pragma unroll
  for (int off = 16; off > 0; off >>= 1) s += __shfl_xor(s, off, 32);
  const float mu = s * (1.f / 256.f);

  float q = 0.f;
#pragma unroll
  for (int j = 0; j < 8; ++j) { const float dv = v[j] - mu; q += dv * dv; }
#pragma unroll
  for (int off = 16; off > 0; off >>= 1) q += __shfl_xor(q, off, 32);
  const float rs = rsqrtf(q * (1.f / 256.f) + 1e-5f);

#pragma unroll
  for (int j = 0; j < 8; ++j) {
    const int c = lane + j * 32;
    float o = (v[j] - mu) * rs * g[c] + bta[c];
    o = o > 0.f ? o : 0.f;
    hidden[base + c] = o;
    h16[base + c] = (half_t)o;
  }
}

// ---------------------------------------------------------------------------
extern "C" void kernel_launch(void* const* d_in, const int* in_sizes, int n_in,
                              void* d_out, int out_size, void* d_ws,
                              size_t ws_size, hipStream_t stream) {
  const float* x    = (const float*)d_in[0];
  const int*   edg  = (const int*)d_in[1];
  const int*   ety  = (const int*)d_in[2];
  const float* Win  = (const float*)d_in[3];
  const float* b_in = (const float*)d_in[4];
  const float* Wq   = (const float*)d_in[5];
  const float* bq   = (const float*)d_in[6];
  const float* Wk   = (const float*)d_in[7];
  const float* bk   = (const float*)d_in[8];
  const float* Wv   = (const float*)d_in[9];
  const float* bv   = (const float*)d_in[10];
  const float* Eemb = (const float*)d_in[11];
  const float* Wo   = (const float*)d_in[12];
  const float* bo   = (const float*)d_in[13];
  const float* lng  = (const float*)d_in[14];
  const float* lnb  = (const float*)d_in[15];
  const float* W1   = (const float*)d_in[16];
  const float* b1   = (const float*)d_in[17];
  const float* W2   = (const float*)d_in[18];
  const float* b2   = (const float*)d_in[19];
  float* out = (float*)d_out;

  constexpr size_t WMAT  = 65536;                 // halves per 256x256 matrix
  constexpr size_t NELEM = (size_t)NNODE * HIDN;  // 5,120,000

  // ---- workspace carve-up (all chunks are multiples of 256 bytes) ----
  char* ws = (char*)d_ws;
  size_t off = 0;
  half_t* w16 = (half_t*)(ws + off); off += 15 * WMAT * sizeof(half_t);
  half_t* x16 = (half_t*)(ws + off); off += NELEM * sizeof(half_t);
  half_t* h16 = (half_t*)(ws + off); off += NELEM * sizeof(half_t);
  half_t* q16 = (half_t*)(ws + off); off += NELEM * sizeof(half_t);
  half_t* k16 = (half_t*)(ws + off); off += NELEM * sizeof(half_t);
  half_t* v16 = (half_t*)(ws + off); off += NELEM * sizeof(half_t);
  half_t* t16 = (half_t*)(ws + off); off += NELEM * sizeof(half_t);
  float* hidden = (float*)(ws + off); off += NELEM * sizeof(float);
  float* agg    = (float*)(ws + off); off += NELEM * sizeof(float);
  (void)ws_size; (void)in_sizes; (void)n_in; (void)out_size;

  half_t* Win16 = w16;
  half_t* Wq16  = w16 + 1 * WMAT;   // [3] matrices
  half_t* Wk16  = w16 + 4 * WMAT;   // [3]
  half_t* Wv16  = w16 + 7 * WMAT;   // [3]
  half_t* Wo16  = w16 + 10 * WMAT;  // [3]
  half_t* W116  = w16 + 13 * WMAT;
  half_t* W216  = w16 + 14 * WMAT;

  const int cb = 256;
  k_f32_to_f16<<<(int)((1 * WMAT + cb - 1) / cb), cb, 0, stream>>>(Win, Win16, (int)(1 * WMAT));
  k_f32_to_f16<<<(int)((3 * WMAT + cb - 1) / cb), cb, 0, stream>>>(Wq, Wq16, (int)(3 * WMAT));
  k_f32_to_f16<<<(int)((3 * WMAT + cb - 1) / cb), cb, 0, stream>>>(Wk, Wk16, (int)(3 * WMAT));
  k_f32_to_f16<<<(int)((3 * WMAT + cb - 1) / cb), cb, 0, stream>>>(Wv, Wv16, (int)(3 * WMAT));
  k_f32_to_f16<<<(int)((3 * WMAT + cb - 1) / cb), cb, 0, stream>>>(Wo, Wo16, (int)(3 * WMAT));
  k_f32_to_f16<<<(int)((1 * WMAT + cb - 1) / cb), cb, 0, stream>>>(W1, W116, (int)(1 * WMAT));
  k_f32_to_f16<<<(int)((1 * WMAT + cb - 1) / cb), cb, 0, stream>>>(W2, W216, (int)(1 * WMAT));
  k_f32_to_f16<<<(int)((NELEM + cb - 1) / cb), cb, 0, stream>>>(x, x16, (int)NELEM);

  const dim3 ggrid(NNODE / 32), gblk(256);  // 625 blocks, 8 waves each

  // hidden = x @ Win^T + b_in  (+ f16 mirror)
  k_gemm<0><<<ggrid, gblk, 0, stream>>>(x16, Win16, b_in, nullptr, hidden, h16);

  for (int l = 0; l < 3; ++l) {
    k_gemm<1><<<ggrid, gblk, 0, stream>>>(h16, Wq16 + l * WMAT, bq + l * HIDN,
                                          nullptr, nullptr, q16);
    k_gemm<1><<<ggrid, gblk, 0, stream>>>(h16, Wk16 + l * WMAT, bk + l * HIDN,
                                          nullptr, nullptr, k16);
    k_gemm<1><<<ggrid, gblk, 0, stream>>>(h16, Wv16 + l * WMAT, bv + l * HIDN,
                                          nullptr, nullptr, v16);
    k_zero4<<<(int)((NELEM / 4 + cb - 1) / cb), cb, 0, stream>>>(agg, (int)NELEM);
    k_edge<<<NEDGE / 16, 128, 0, stream>>>(q16, k16, v16, edg, ety,
                                           Eemb + (size_t)l * 3 * HIDN,
                                           Wo16 + l * WMAT, bo + l * HIDN, agg);
    k_ln_relu<<<NNODE / 8, 256, 0, stream>>>(agg, lng + l * HIDN,
                                             lnb + l * HIDN, hidden, h16);
  }

  // adjustment = relu(h @ W1^T + b1) @ W2^T + b2 ; out = x + adjustment
  k_gemm<2><<<ggrid, gblk, 0, stream>>>(h16, W116, b1, nullptr, nullptr, t16);
  k_gemm<3><<<ggrid, gblk, 0, stream>>>(t16, W216, b2, x, out, nullptr);
}
